// GroupedMLP_9328668967830
// MI455X (gfx1250) — matrix operand order repaired and checked by reading, not compiled, not run on hardware
//
#include <hip/hip_runtime.h>
#include <hip/hip_bf16.h>

#define E_ 8
#define T_ 1024
#define H_ 2048
#define I_ 5632

typedef __bf16 bf16x16 __attribute__((ext_vector_type(16)));
typedef float  f32x8   __attribute__((ext_vector_type(8)));
typedef int    v4i     __attribute__((vector_size(16)));   // matches builtin param

#if defined(__has_builtin)
#  if __has_builtin(__builtin_amdgcn_global_load_async_to_lds_b128)
#    define HAS_ASYNC_LDS 1
#  endif
#  if __has_builtin(__builtin_amdgcn_s_wait_asynccnt)
#    define HAS_WAIT_ASYNC 1
#  endif
#endif
#ifndef HAS_ASYNC_LDS
#  define HAS_ASYNC_LDS 0
#endif
#ifndef HAS_WAIT_ASYNC
#  define HAS_WAIT_ASYNC 0
#endif

union Frag {
  bf16x16 h;
  uint4   q[2];
};

__device__ __forceinline__ unsigned short f32_to_bf16(float f) {
  return (unsigned short)(__float_as_uint(f) >> 16);  // truncate mantissa
}

__device__ __forceinline__ void wait_async_lds() {
#if HAS_WAIT_ASYNC
  __builtin_amdgcn_s_wait_asynccnt(0);
#else
  asm volatile("s_wait_asynccnt 0x0" ::: "memory");
#endif
}

// ---------------------------------------------------------------------------
// Kernel 1: fc1 + GLU.  Per block: 128 token rows x 64 intermediate columns.
// C_a = X @ W1[:, n0:n0+64], C_b = X @ W1[:, I+n0:I+n0+64],
// inter = silu(C_a)*C_b  (bf16, workspace).
// ---------------------------------------------------------------------------
__global__ __launch_bounds__(256) void fc1_glu_kernel(
    const float* __restrict__ x, const float* __restrict__ w1,
    unsigned short* __restrict__ inter) {
  __shared__ unsigned short sA [128 * 40];
  __shared__ unsigned short sBa[ 64 * 40];
  __shared__ unsigned short sBb[ 64 * 40];

  const int tid  = threadIdx.x;
  const int nblk = blockIdx.x;           // fast dim: same expert together (L2)
  const int mblk = blockIdx.y;
  const int row0 = mblk * 128;
  const int e    = row0 / T_;
  const int col0 = nblk * 64;
  const float* w1e = w1 + (size_t)e * H_ * (2 * I_);

  const int lane   = tid & 31;
  const int waveId = tid >> 5;
  const int wm   = waveId & 3;           // 32-row chunk
  const int wn   = waveId >> 2;          // 32-col chunk
  const int hsel = lane >> 4;
  const int l15  = lane & 15;

  f32x8 acc_a[2][2], acc_b[2][2];
  const f32x8 zero = {0.f,0.f,0.f,0.f,0.f,0.f,0.f,0.f};
#pragma unroll
  for (int f = 0; f < 2; ++f)
#pragma unroll
    for (int g = 0; g < 2; ++g) { acc_a[f][g] = zero; acc_b[f][g] = zero; }

  for (int kk = 0; kk < H_ / 32; ++kk) {
    const int k0 = kk * 32;

    // ---- stage A: 128x32 fp32 -> bf16, row-major ----
#pragma unroll
    for (int p = 0; p < 4; ++p) {
      int idx = tid + p * 256;
      int row = idx >> 3;
      int k   = (idx & 7) * 4;
      float4 v = *(const float4*)(x + (size_t)(row0 + row) * H_ + k0 + k);
      uint2 u;
      u.x = ((unsigned)f32_to_bf16(v.y) << 16) | f32_to_bf16(v.x);
      u.y = ((unsigned)f32_to_bf16(v.w) << 16) | f32_to_bf16(v.z);
      *(uint2*)&sA[row * 40 + k] = u;
    }
    // ---- stage B_a / B_b: 32x64 fp32, transposed to [n][k] bf16 ----
#pragma unroll
    for (int p = 0; p < 2; ++p) {
      int idx = tid + p * 256;
      int k = idx >> 4;                  // 0..31
      int n = (idx & 15) * 4;            // 0..60
      const float* pa = w1e + (size_t)(k0 + k) * (2 * I_) + col0 + n;
      float4 va = *(const float4*)pa;
      float4 vb = *(const float4*)(pa + I_);
      sBa[(n + 0) * 40 + k] = f32_to_bf16(va.x);
      sBa[(n + 1) * 40 + k] = f32_to_bf16(va.y);
      sBa[(n + 2) * 40 + k] = f32_to_bf16(va.z);
      sBa[(n + 3) * 40 + k] = f32_to_bf16(va.w);
      sBb[(n + 0) * 40 + k] = f32_to_bf16(vb.x);
      sBb[(n + 1) * 40 + k] = f32_to_bf16(vb.y);
      sBb[(n + 2) * 40 + k] = f32_to_bf16(vb.z);
      sBb[(n + 3) * 40 + k] = f32_to_bf16(vb.w);
    }
    // hint the next K-tile toward L2 while WMMAs run (global_prefetch_b8)
    if (kk + 1 < H_ / 32) {
      __builtin_prefetch(x + (size_t)(row0 + (tid >> 3)) * H_ + k0 + 32, 0, 0);
      __builtin_prefetch(w1e + (size_t)(k0 + 32 + (tid >> 4)) * (2 * I_) + col0, 0, 0);
    }
    __syncthreads();

    // ---- fragments (layouts per CDNA5 ISA 7.12.2) ----
    Frag fa[2], fba[2], fbb[2];
#pragma unroll
    for (int f = 0; f < 2; ++f) {
      const char* base = (const char*)sA + (wm * 32 + f * 16 + l15) * 80 + hsel * 16;
      fa[f].q[0] = *(const uint4*)(base);
      fa[f].q[1] = *(const uint4*)(base + 32);
    }
#pragma unroll
    for (int g = 0; g < 2; ++g) {
      int cb = (wn * 32 + g * 16 + l15) * 80 + hsel * 16;
      fba[g].q[0] = *(const uint4*)((const char*)sBa + cb);
      fba[g].q[1] = *(const uint4*)((const char*)sBa + cb + 32);
      fbb[g].q[0] = *(const uint4*)((const char*)sBb + cb);
      fbb[g].q[1] = *(const uint4*)((const char*)sBb + cb + 32);
    }
#pragma unroll
    for (int f = 0; f < 2; ++f)
#pragma unroll
      for (int g = 0; g < 2; ++g) {
        acc_a[f][g] = __builtin_amdgcn_wmma_f32_16x16x32_bf16(
            false, fa[f].h, false, fba[g].h, (short)0, acc_a[f][g], false, false);
        acc_b[f][g] = __builtin_amdgcn_wmma_f32_16x16x32_bf16(
            false, fa[f].h, false, fbb[g].h, (short)0, acc_b[f][g], false, false);
      }
    __syncthreads();
  }

  // ---- GLU epilogue: silu(a)*b via v_exp_f32 + v_rcp_f32 (no IEEE div) ----
  const int i0 = col0 + wn * 32;
#pragma unroll
  for (int f = 0; f < 2; ++f)
#pragma unroll
    for (int g = 0; g < 2; ++g)
#pragma unroll
      for (int r = 0; r < 8; ++r) {
        int m = wm * 32 + f * 16 + hsel * 8 + r;
        int n = i0 + g * 16 + l15;
        float a = acc_a[f][g][r];
        float b = acc_b[f][g][r];
        float h = a * __builtin_amdgcn_rcpf(1.0f + __expf(-a)) * b;
        inter[(size_t)(row0 + m) * I_ + n] = f32_to_bf16(h);
      }
}

// ---------------------------------------------------------------------------
// Kernel 2: out = inter(bf16) @ W2.  Per block: 128 rows x 128 cols, K = I.
// A tile is a pure bf16 copy -> use GLOBAL_LOAD_ASYNC_TO_LDS (ASYNCcnt path).
// ---------------------------------------------------------------------------
__global__ __launch_bounds__(256) void fc2_kernel(
    const unsigned short* __restrict__ inter, const float* __restrict__ w2,
    float* __restrict__ out) {
  __shared__ unsigned short sA[128 * 40];
  __shared__ unsigned short sB[128 * 40];   // transposed [n][k]

  const int tid   = threadIdx.x;
  const int ncol0 = blockIdx.x * 128;
  const int mblk  = blockIdx.y;
  const int row0  = mblk * 128;
  const int e     = row0 / T_;
  const float* w2e = w2 + (size_t)e * I_ * H_;

  const int lane   = tid & 31;
  const int waveId = tid >> 5;
  const int wm   = waveId & 3;              // 32-row chunk
  const int wn   = waveId >> 2;             // 64-col chunk
  const int hsel = lane >> 4;
  const int l15  = lane & 15;

  f32x8 acc[2][4];
  const f32x8 zero = {0.f,0.f,0.f,0.f,0.f,0.f,0.f,0.f};
#pragma unroll
  for (int f = 0; f < 2; ++f)
#pragma unroll
    for (int g = 0; g < 4; ++g) acc[f][g] = zero;

  for (int kk = 0; kk < I_ / 32; ++kk) {
    const int k0 = kk * 32;

    // ---- stage A: 128x32 halves, async DMA global->LDS (no VGPR bounce) ----
#pragma unroll
    for (int p = 0; p < 2; ++p) {
      int idx = tid + p * 256;
      int row = idx >> 2;
      int kq  = (idx & 3) * 8;              // 8 halves = 16B per slot
      const unsigned short* g = inter + (size_t)(row0 + row) * I_ + k0 + kq;
      unsigned short*       l = &sA[row * 40 + kq];
#if HAS_ASYNC_LDS
      __builtin_amdgcn_global_load_async_to_lds_b128((v4i*)g, (v4i*)l, 0, 0);
#else
      *(uint4*)l = *(const uint4*)g;
#endif
    }
    // ---- stage B: 32x128 fp32 -> bf16 transposed ----
#pragma unroll
    for (int p = 0; p < 4; ++p) {
      int idx = tid + p * 256;
      int k = idx >> 5;                     // 0..31
      int n = (idx & 31) * 4;               // 0..124
      float4 v = *(const float4*)(w2e + (size_t)(k0 + k) * H_ + ncol0 + n);
      sB[(n + 0) * 40 + k] = f32_to_bf16(v.x);
      sB[(n + 1) * 40 + k] = f32_to_bf16(v.y);
      sB[(n + 2) * 40 + k] = f32_to_bf16(v.z);
      sB[(n + 3) * 40 + k] = f32_to_bf16(v.w);
    }
    if (kk + 1 < I_ / 32) {
      __builtin_prefetch(inter + (size_t)(row0 + (tid >> 2)) * I_ + k0 + 32, 0, 0);
      __builtin_prefetch(w2e + (size_t)(k0 + 32 + (tid >> 5)) * H_ + ncol0, 0, 0);
    }
#if HAS_ASYNC_LDS
    wait_async_lds();                       // LDS writes visible before barrier
#endif
    __syncthreads();

    Frag fa[2], fb[4];
#pragma unroll
    for (int f = 0; f < 2; ++f) {
      const char* base = (const char*)sA + (wm * 32 + f * 16 + l15) * 80 + hsel * 16;
      fa[f].q[0] = *(const uint4*)(base);
      fa[f].q[1] = *(const uint4*)(base + 32);
    }
#pragma unroll
    for (int g = 0; g < 4; ++g) {
      int cb = (wn * 64 + g * 16 + l15) * 80 + hsel * 16;
      fb[g].q[0] = *(const uint4*)((const char*)sB + cb);
      fb[g].q[1] = *(const uint4*)((const char*)sB + cb + 32);
    }
#pragma unroll
    for (int f = 0; f < 2; ++f)
#pragma unroll
      for (int g = 0; g < 4; ++g)
        acc[f][g] = __builtin_amdgcn_wmma_f32_16x16x32_bf16(
            false, fa[f].h, false, fb[g].h, (short)0, acc[f][g], false, false);
    __syncthreads();
  }

  // ---- fp32 epilogue: half-wave contiguous 64B stores ----
#pragma unroll
  for (int f = 0; f < 2; ++f)
#pragma unroll
    for (int g = 0; g < 4; ++g)
#pragma unroll
      for (int r = 0; r < 8; ++r) {
        int m = wm * 32 + f * 16 + hsel * 8 + r;
        int n = ncol0 + wn * 64 + g * 16 + l15;
        out[(size_t)(row0 + m) * H_ + n] = acc[f][g][r];
      }
}

extern "C" void kernel_launch(void* const* d_in, const int* in_sizes, int n_in,
                              void* d_out, int out_size, void* d_ws, size_t ws_size,
                              hipStream_t stream) {
  (void)in_sizes; (void)n_in; (void)out_size; (void)ws_size;
  const float* x  = (const float*)d_in[0];
  const float* w1 = (const float*)d_in[1];
  const float* w2 = (const float*)d_in[2];
  float* out = (float*)d_out;
  unsigned short* inter = (unsigned short*)d_ws;  // E*T*I bf16 = 92.3 MB

  fc1_glu_kernel<<<dim3(I_ / 64, (E_ * T_) / 128), 256, 0, stream>>>(x, w1, inter);
  fc2_kernel<<<dim3(H_ / 128, (E_ * T_) / 128), 256, 0, stream>>>(inter, w2, out);
}
